// MultiHeadTrendAwareAttention_1391569404154
// MI455X (gfx1250) — matrix operand order repaired and checked by reading, not compiled, not run on hardware
//
#include <hip/hip_runtime.h>
#include <hip/hip_bf16.h>

// ---------- constants ----------
#define B_  4
#define N_  2048
#define D_  512
#define H_  8
#define HD_ 64
#define KS_ 3
#define NP_ (N_ + 2)          // padded sequence rows per batch

typedef unsigned short u16;
typedef __bf16 v16bf __attribute__((ext_vector_type(16)));
typedef float  v8f   __attribute__((ext_vector_type(8)));
typedef unsigned int v4u __attribute__((ext_vector_type(4)));

union FragU { v16bf v; v4u u[2]; u16 h[16]; };

__device__ __forceinline__ u16 f32_to_bf16(float f) {
    union { float f; unsigned u; } v; v.f = f;
    unsigned r = v.u + 0x7FFFu + ((v.u >> 16) & 1u);
    return (u16)(r >> 16);
}

__device__ __forceinline__ v8f wmma_bf16(v16bf a, v16bf b, v8f c) {
    return __builtin_amdgcn_wmma_f32_16x16x32_bf16(false, a, false, b, (short)0, c,
                                                   false, false);
}

// Async global -> LDS copy, 16B per lane, tracked by ASYNCcnt (CDNA5 TDM-lite path).
// LDS operand: wave-relative offset = low 32 bits of the generic shared pointer
// (ISA 10.2: LDS aperture address = addr[31:0]).
__device__ __forceinline__ void async_load_b128(const u16* lds_dst, const u16* gsrc) {
    unsigned loff = (unsigned)(unsigned long long)lds_dst;
    unsigned long long g = (unsigned long long)gsrc;
    asm volatile("global_load_async_to_lds_b128 %0, %1, off"
                 :: "v"(loff), "v"(g) : "memory");
}
#define WAIT_ASYNC(n) asm volatile("s_wait_asynccnt %0" :: "i"(n) : "memory")

// A fragment (16x32 bf16) from LDS tile, row stride ldh halves (multiple of 8).
// lane l: m = mbase + l%16, kg = (l/16)*8; halves 0..7 -> K=koff+kg+h,
// halves 8..15 -> K=koff+16+kg+(h-8). Two 16B ds reads.
__device__ __forceinline__ v16bf frag_A(const u16* tile, int ldh, int mbase, int koff) {
    int lane = threadIdx.x & 31;
    const u16* p = tile + (mbase + (lane & 15)) * ldh + koff + ((lane >> 4) << 3);
    FragU f;
    f.u[0] = *(const v4u*)(p);
    f.u[1] = *(const v4u*)(p + 16);
    return f.v;
}

// B fragment from Bt LDS tile stored [n][k] row-major.
// lane l: n = nbase + l%16, kg = (l/16)*8; halves 0..7 -> K=koff+kg+h, halves 8..15 -> K=koff+16+kg+(h-8).
__device__ __forceinline__ v16bf frag_B(const u16* tile, int ldh, int nbase, int koff) {
    int lane = threadIdx.x & 31;
    const u16* p = tile + (nbase + (lane & 15)) * ldh + koff + ((lane >> 4) << 3);
    FragU f;
    f.u[0] = *(const v4u*)(p);
    f.u[1] = *(const v4u*)(p + 16);
    return f.v;
}

// ---------- LayerNorm + bf16 convert into sequence-padded buffer ----------
__global__ __launch_bounds__(256)
void ln_kernel(const float* __restrict__ x, const float* __restrict__ gamma,
               const float* __restrict__ beta, u16* __restrict__ xnp) {
    const int pr = blockIdx.x;               // 0 .. B*(N+2)-1
    const int b = pr / NP_, r = pr % NP_;
    u16* outrow = xnp + (long long)pr * D_;
    const int t = threadIdx.x;
    if (r == 0 || r == NP_ - 1) {            // zero pad rows
        outrow[t] = 0; outrow[t + 256] = 0;
        return;
    }
    const float* xrow = x + ((long long)b * N_ + (r - 1)) * D_;
    float v0 = xrow[t], v1 = xrow[t + 256];
    __shared__ float red[256];
    red[t] = v0 + v1;
    __syncthreads();
    for (int o = 128; o > 0; o >>= 1) { if (t < o) red[t] += red[t + o]; __syncthreads(); }
    float mu = red[0] * (1.0f / D_);
    __syncthreads();
    float d0 = v0 - mu, d1 = v1 - mu;
    red[t] = d0 * d0 + d1 * d1;
    __syncthreads();
    for (int o = 128; o > 0; o >>= 1) { if (t < o) red[t] += red[t + o]; __syncthreads(); }
    float rstd = rsqrtf(red[0] * (1.0f / D_) + 1e-5f);
    outrow[t]       = f32_to_bf16(d0 * rstd * gamma[t]       + beta[t]);
    outrow[t + 256] = f32_to_bf16(d1 * rstd * gamma[t + 256] + beta[t + 256]);
}

// ---------- weight repacks ----------
// conv W: [dout][din][3] fp32 -> Bt bf16 [dout][k*D+din]
__global__ __launch_bounds__(256)
void conv_w_kernel(const float* __restrict__ W, u16* __restrict__ out) {
    int i = blockIdx.x * 256 + threadIdx.x;
    if (i >= D_ * D_ * KS_) return;
    int dout = i / (D_ * KS_);
    int rem  = i % (D_ * KS_);
    int k = rem / D_, din = rem % D_;
    out[i] = f32_to_bf16(W[((long long)dout * D_ + din) * KS_ + k]);
}
// linear W (out,in) fp32 -> bf16 (already Bt layout)
__global__ __launch_bounds__(256)
void lin_w_kernel(const float* __restrict__ W, u16* __restrict__ out, int n) {
    int i = blockIdx.x * 256 + threadIdx.x;
    if (i < n) out[i] = f32_to_bf16(W[i]);
}

// ---------- generic bf16 GEMM:  C = A @ Bt^T + bias ----------
// A: row-major bf16, element (m,kk) at A[(m + kk/lda)*lda + kk%lda]  (conv fold)
// Bt: row-major [Nn][K] bf16. WG tile 128x64, 8 waves of 32x32 each.
// K chunk = 64, double-buffered LDS, async global->LDS staging.
template <bool OUT_BF16>
__global__ __launch_bounds__(256)
void gemm_bt_kernel(const u16* __restrict__ A, const u16* __restrict__ Bt,
                    const float* __restrict__ bias, void* __restrict__ Cv,
                    int Kd, int lda, int ldb, int ldc,
                    long long strideA, long long strideC) {
    __shared__ u16 As[2][128 * 72];
    __shared__ u16 Bs[2][64 * 72];
    const int t = threadIdx.x;
    const int wave = t >> 5, lane = t & 31;
    const int m0 = blockIdx.y * 128;
    const int n0 = blockIdx.x * 64;
    A += (long long)blockIdx.z * strideA;
    const int wm = (wave >> 1) * 32;
    const int wn = (wave & 1) * 32;
    v8f c[2][2] = {};

    auto issue = [&](int k0, int buf) {       // 6 async b128 per thread per chunk
        const int arow = m0 + k0 / lda;       // conv tap folded into row offset
        const int acol = k0 % lda;
#pragma unroll
        for (int i = 0; i < 4; ++i) {         // A tile 128x64
            int v = t + i * 256;
            int r = v >> 3, c8 = (v & 7) << 3;
            async_load_b128(&As[buf][r * 72 + c8],
                            A + (long long)(arow + r) * lda + acol + c8);
        }
#pragma unroll
        for (int i = 0; i < 2; ++i) {         // Bt tile 64x64
            int v = t + i * 256;
            int r = v >> 3, c8 = (v & 7) << 3;
            async_load_b128(&Bs[buf][r * 72 + c8],
                            Bt + (long long)(n0 + r) * ldb + k0 + c8);
        }
    };

    issue(0, 0);
    const int nch = Kd >> 6;
    for (int i = 0; i < nch; ++i) {
        if (i + 1 < nch) {                    // overlap copy of next chunk
            issue((i + 1) << 6, (i + 1) & 1);
            WAIT_ASYNC(6);                    // chunk i complete, next 6 in flight
        } else {
            WAIT_ASYNC(0);
        }
        __syncthreads();
        const u16* as = As[i & 1];
        const u16* bs = Bs[i & 1];
#pragma unroll
        for (int ks = 0; ks < 2; ++ks) {
            v16bf a0 = frag_A(as, 72, wm + 0,  ks * 32);
            v16bf a1 = frag_A(as, 72, wm + 16, ks * 32);
            v16bf b0 = frag_B(bs, 72, wn + 0,  ks * 32);
            v16bf b1 = frag_B(bs, 72, wn + 16, ks * 32);
            c[0][0] = wmma_bf16(a0, b0, c[0][0]);
            c[0][1] = wmma_bf16(a0, b1, c[0][1]);
            c[1][0] = wmma_bf16(a1, b0, c[1][0]);
            c[1][1] = wmma_bf16(a1, b1, c[1][1]);
        }
        __syncthreads();
    }
    const int lrow = (lane >> 4) << 3;        // C layout: VGPR i -> row i or 8+i
    const int lcol = lane & 15;
#pragma unroll
    for (int mi = 0; mi < 2; ++mi)
#pragma unroll
        for (int ni = 0; ni < 2; ++ni) {
            int gm = m0 + wm + mi * 16 + lrow;
            int gn = n0 + wn + ni * 16 + lcol;
            float bv = bias[gn];
#pragma unroll
            for (int i = 0; i < 8; ++i) {
                float val = c[mi][ni][i] + bv;
                long long off = (long long)(gm + i) * ldc + gn +
                                (long long)blockIdx.z * strideC;
                if (OUT_BF16) ((u16*)Cv)[off] = f32_to_bf16(val);
                else          ((float*)Cv)[off] = val;
            }
        }
}

// ---------- flash attention (per b,h; 128-query tile; 32-key steps) ----------
__global__ __launch_bounds__(256)
void attn_kernel(const u16* __restrict__ Q, const u16* __restrict__ K,
                 const u16* __restrict__ V, u16* __restrict__ AO) {
    __shared__ u16  Qs[128 * 72];
    __shared__ u16  Ks[32 * 72];
    __shared__ u16  VTs[64 * 40];
    __shared__ u16  Ps[128 * 40];
    __shared__ float Ss[128 * 33];
    __shared__ float mbuf[128], lbuf[128], abuf[128];
    const int t = threadIdx.x;
    const int wave = t >> 5, lane = t & 31;
    const int b = blockIdx.y >> 3, h = blockIdx.y & 7;
    const int q0 = blockIdx.x * 128;
    const long long qbase = ((long long)b * N_ + q0) * D_ + h * HD_;
#pragma unroll
    for (int i = 0; i < 4; ++i) {             // Q tile 128x64, async -> LDS
        int v = t + i * 256;
        int r = v >> 3, c8 = (v & 7) << 3;
        async_load_b128(&Qs[r * 72 + c8], Q + qbase + (long long)r * D_ + c8);
    }
    if (t < 128) { mbuf[t] = -1e30f; lbuf[t] = 0.f; }

    const int qm = wave * 16;
    const int lrow = (lane >> 4) << 3;
    const int lcol = lane & 15;
    v8f oc[4] = {};                           // per-wave O accumulator 16x64

    for (int k0 = 0; k0 < N_; k0 += 32) {
        const long long kbase = ((long long)b * N_ + k0) * D_ + h * HD_;
        {                                     // K tile 32x64 async; V^T via VGPRs
            int r = t >> 3, c8 = (t & 7) << 3;
            async_load_b128(&Ks[r * 72 + c8], K + kbase + (long long)r * D_ + c8);
            union { v4u u; u16 s[8]; } vv;
            vv.u = *(const v4u*)(V + kbase + (long long)r * D_ + c8);
#pragma unroll
            for (int j = 0; j < 8; ++j) VTs[(c8 + j) * 40 + r] = vv.s[j];
        }
        WAIT_ASYNC(0);                        // Q (first iter) + K landed in LDS
        __syncthreads();
        {                                     // S = Q K^T  (K-dim 64 = 2 steps)
            v16bf aq0 = frag_A(Qs, 72, qm, 0);
            v16bf aq1 = frag_A(Qs, 72, qm, 32);
#pragma unroll
            for (int nt = 0; nt < 2; ++nt) {
                v8f s = {};
                s = wmma_bf16(aq0, frag_B(Ks, 72, nt * 16, 0),  s);
                s = wmma_bf16(aq1, frag_B(Ks, 72, nt * 16, 32), s);
                int gn = nt * 16 + lcol;
#pragma unroll
                for (int i = 0; i < 8; ++i) Ss[(qm + lrow + i) * 33 + gn] = s[i];
            }
        }
        __syncthreads();
        if (t < 128) {                        // online softmax, row t
            const float scale = 0.125f;       // HD^-0.5, exact
            float mold = mbuf[t], mt = mold;
            float sv[32];
#pragma unroll
            for (int j = 0; j < 32; ++j) {
                sv[j] = Ss[t * 33 + j] * scale;
                mt = fmaxf(mt, sv[j]);
            }
            float alpha = __expf(mold - mt);
            float lsum = 0.f;
#pragma unroll
            for (int j = 0; j < 32; ++j) {
                float p = __expf(sv[j] - mt);
                lsum += p;
                Ps[t * 40 + j] = f32_to_bf16(p);
            }
            mbuf[t] = mt;
            lbuf[t] = lbuf[t] * alpha + lsum;
            abuf[t] = alpha;
        }
        __syncthreads();
        {                                     // O = alpha*O + P V
            v16bf ap = frag_A(Ps, 40, qm, 0);
            float al[8];
#pragma unroll
            for (int i = 0; i < 8; ++i) al[i] = abuf[qm + lrow + i];
#pragma unroll
            for (int nt = 0; nt < 4; ++nt) {
#pragma unroll
                for (int i = 0; i < 8; ++i) oc[nt][i] *= al[i];
                oc[nt] = wmma_bf16(ap, frag_B(VTs, 40, nt * 16, 0), oc[nt]);
            }
        }
        __syncthreads();
    }
    float linv[8];
#pragma unroll
    for (int i = 0; i < 8; ++i) linv[i] = 1.0f / lbuf[qm + lrow + i];
#pragma unroll
    for (int nt = 0; nt < 4; ++nt) {
        int gn = nt * 16 + lcol;
#pragma unroll
        for (int i = 0; i < 8; ++i) {
            int gr = qm + lrow + i;
            AO[qbase + (long long)gr * D_ + gn] = f32_to_bf16(oc[nt][i] * linv[i]);
        }
    }
}

// ---------- host launch ----------
extern "C" void kernel_launch(void* const* d_in, const int* in_sizes, int n_in,
                              void* d_out, int out_size, void* d_ws, size_t ws_size,
                              hipStream_t stream) {
    const float* x     = (const float*)d_in[0];
    const float* gamma = (const float*)d_in[1];
    const float* beta  = (const float*)d_in[2];
    const float* Wq    = (const float*)d_in[3];
    const float* bq    = (const float*)d_in[4];
    const float* Wk    = (const float*)d_in[5];
    const float* bk    = (const float*)d_in[6];
    const float* Wv    = (const float*)d_in[7];
    const float* bv    = (const float*)d_in[8];
    const float* Wo    = (const float*)d_in[9];
    const float* bo    = (const float*)d_in[10];

    char* ws = (char*)d_ws;
    size_t off = 0;
    auto alloc = [&](size_t bytes) { char* p = ws + off; off += (bytes + 255) & ~(size_t)255; return p; };
    u16* xnp  = (u16*)alloc((size_t)B_ * NP_ * D_ * 2);   // padded LN output (bf16)
    u16* Qb   = (u16*)alloc((size_t)B_ * N_ * D_ * 2);
    u16* Kb   = (u16*)alloc((size_t)B_ * N_ * D_ * 2);
    u16* Vb   = (u16*)alloc((size_t)B_ * N_ * D_ * 2);
    u16* AOb  = (u16*)alloc((size_t)B_ * N_ * D_ * 2);
    u16* WqT  = (u16*)alloc((size_t)D_ * D_ * KS_ * 2);
    u16* WkT  = (u16*)alloc((size_t)D_ * D_ * KS_ * 2);
    u16* WvT  = (u16*)alloc((size_t)D_ * D_ * 2);
    u16* WoT  = (u16*)alloc((size_t)D_ * D_ * 2);
    (void)ws_size; (void)in_sizes; (void)n_in; (void)out_size;

    // 1) LayerNorm -> padded bf16
    ln_kernel<<<B_ * NP_, 256, 0, stream>>>(x, gamma, beta, xnp);
    // 2) weight repacks
    conv_w_kernel<<<(D_ * D_ * KS_ + 255) / 256, 256, 0, stream>>>(Wq, WqT);
    conv_w_kernel<<<(D_ * D_ * KS_ + 255) / 256, 256, 0, stream>>>(Wk, WkT);
    lin_w_kernel<<<(D_ * D_ + 255) / 256, 256, 0, stream>>>(Wv, WvT, D_ * D_);
    lin_w_kernel<<<(D_ * D_ + 255) / 256, 256, 0, stream>>>(Wo, WoT, D_ * D_);

    const long long sAp = (long long)NP_ * D_;   // padded batch stride (elements)
    const long long sC  = (long long)N_ * D_;
    dim3 ggrid(D_ / 64, N_ / 128, B_);
    // 3) Q,K via conv-as-GEMM (K=1536, tap folded via lda), V via linear
    gemm_bt_kernel<true><<<ggrid, 256, 0, stream>>>(xnp,        WqT, bq, Qb,
                                                    D_ * KS_, D_, D_ * KS_, D_, sAp, sC);
    gemm_bt_kernel<true><<<ggrid, 256, 0, stream>>>(xnp,        WkT, bk, Kb,
                                                    D_ * KS_, D_, D_ * KS_, D_, sAp, sC);
    gemm_bt_kernel<true><<<ggrid, 256, 0, stream>>>(xnp + D_,   WvT, bv, Vb,
                                                    D_, D_, D_, D_, sAp, sC);
    // 4) attention
    attn_kernel<<<dim3(N_ / 128, B_ * H_), 256, 0, stream>>>(Qb, Kb, Vb, AOb);
    // 5) output projection -> d_out (fp32)
    gemm_bt_kernel<false><<<ggrid, 256, 0, stream>>>(AOb, WoT, bo, (float*)d_out,
                                                     D_, D_, D_, D_, sC, sC);
}
